// XFeatModel_35725537968172
// MI455X (gfx1250) — matrix operand order repaired and checked by reading, not compiled, not run on hardware
//
#include <hip/hip_runtime.h>
#include <hip/hip_bf16.h>
#include <math.h>

// ---------------------------------------------------------------------------
// CDNA5 (gfx1250) XFeat matcher: flash-attention correlation + implicit-GEMM
// convs on v_wmma_f32_16x16x32_bf16 (wave32). Vectorized LDS (b128) paths,
// per-lane online softmax (no cross-lane ops in hot loop).
// ---------------------------------------------------------------------------

typedef __attribute__((ext_vector_type(16))) __bf16        v16bf;
typedef __attribute__((ext_vector_type(8)))  float         v8f;
typedef __attribute__((ext_vector_type(4)))  unsigned int  u32x4;
typedef __attribute__((ext_vector_type(2)))  unsigned int  u32x2;

#define HC 64
#define WC 64
#define MC (HC * WC)      // 4096 coarse pixels
#define CC 64             // coarse channels
#define HF 128
#define WF 128
#define MF (HF * WF)      // 16384 fine pixels
#define CF 24             // fine channels
#define NB 4              // batch

__device__ __forceinline__ unsigned short f2bfbits(float f) {
  unsigned u = __builtin_bit_cast(unsigned, f);
  u += 0x7FFFu + ((u >> 16) & 1u);            // round-to-nearest-even
  return (unsigned short)(u >> 16);
}
__device__ __forceinline__ float bf2f(unsigned short s) {
  unsigned u = ((unsigned)s) << 16;
  return __builtin_bit_cast(float, u);
}
__device__ __forceinline__ __bf16 bfbits(unsigned short s) {
  return __builtin_bit_cast(__bf16, s);
}
__device__ __forceinline__ int iclamp(int v, int lo, int hi) {
  return v < lo ? lo : (v > hi ? hi : v);
}

// 32B contiguous LDS -> bf16 fragment (2x ds_load_b128).
struct Frag32B { u32x4 lo, hi; };
__device__ __forceinline__ v16bf load_frag16(const unsigned short* p) {
  Frag32B r;
  r.lo = *(const u32x4*)(p);
  r.hi = *(const u32x4*)(p + 8);
  return __builtin_bit_cast(v16bf, r);
}

// 16-bit A-matrix 16x32 per-lane K map (ISA 7.12.2): element e, lane half hi.
__device__ __forceinline__ int a_kmap(int e, int laneHi) {
  return (e & 7) | ((e >> 3) << 4) | (laneHi << 3);
}
// inverse: storage position within a 32-K row so that element e is contiguous
__device__ __forceinline__ int a_kpos(int k) {
  return (((k >> 3) & 1) << 4) | (k & 7) | ((k >> 4) << 3);
}

// ---------------------------------------------------------------------------
// Kernel 1: fused correlation + online softmax + P@grid (flash attention).
// Q=feats0_c (B,64,4096)/8, K=feats1_c, V=pos grid. 4 waves/block, 16 rows/wave.
// K-tile (64k x 16n bf16) double-buffered in LDS, transposed [n][k] (pad 72)
// so B fragments are contiguous 32B loads. Softmax state is per-lane; single
// exact merge at the end (no cross-lane ops in the loop).
// ---------------------------------------------------------------------------
__global__ __launch_bounds__(128) void corr_softmax_warp_kernel(
    const float* __restrict__ f0, const float* __restrict__ f1,
    float* __restrict__ warp /* (B,2,MC) */) {
  __shared__ unsigned short Ks[2][16 * 72];   // [n][k] rows padded to 72
  const int b      = blockIdx.y;
  const int lane   = threadIdx.x & 31;
  const int wave   = threadIdx.x >> 5;
  const int laneHi = lane >> 4;
  const int lane16 = lane & 15;
  const int m0     = blockIdx.x * 64 + wave * 16;
  const float* f0b = f0 + (size_t)b * CC * MC;
  const float* f1b = f1 + (size_t)b * CC * MC;

  // staging geometry: thread owns (n = tid&15, k = (tid>>4)*8 .. +7)
  const int sn = threadIdx.x & 15;
  const int sk = (threadIdx.x >> 4) * 8;
  const float* sg = f1b + (size_t)sk * MC + sn;

  // Q fragments (16 rows x 64 K), 1/sqrt(C)=0.125 folded; loaded once.
  v16bf a0, a1;
  const int mrowA = m0 + lane16;
#pragma unroll
  for (int e = 0; e < 16; ++e) {
    int k = laneHi * 16 + e;
    a0[e] = bfbits(f2bfbits(f0b[(size_t)k * MC + mrowA] * 0.125f));
    a1[e] = bfbits(f2bfbits(f0b[(size_t)(k + 32) * MC + mrowA] * 0.125f));
  }

  // stage one 64x16 K-tile into buffer `buf` as [n][k], packed b128 stores
  auto stage = [&](int buf, int n0) {
    unsigned short t0 = f2bfbits(sg[(size_t)0 * MC + n0]);
    unsigned short t1 = f2bfbits(sg[(size_t)1 * MC + n0]);
    unsigned short t2 = f2bfbits(sg[(size_t)2 * MC + n0]);
    unsigned short t3 = f2bfbits(sg[(size_t)3 * MC + n0]);
    unsigned short t4 = f2bfbits(sg[(size_t)4 * MC + n0]);
    unsigned short t5 = f2bfbits(sg[(size_t)5 * MC + n0]);
    unsigned short t6 = f2bfbits(sg[(size_t)6 * MC + n0]);
    unsigned short t7 = f2bfbits(sg[(size_t)7 * MC + n0]);
    u32x4 pk;
    pk.x = (unsigned)t0 | ((unsigned)t1 << 16);
    pk.y = (unsigned)t2 | ((unsigned)t3 << 16);
    pk.z = (unsigned)t4 | ((unsigned)t5 << 16);
    pk.w = (unsigned)t6 | ((unsigned)t7 << 16);
    *(u32x4*)&Ks[buf][sn * 72 + sk] = pk;
  };

  float mrow[8], lrow[8], ax[8], ay[8];
#pragma unroll
  for (int r = 0; r < 8; ++r) {
    mrow[r] = -__builtin_inff(); lrow[r] = 0.f; ax[r] = 0.f; ay[r] = 0.f;
  }

  stage(0, 0);
  const int NIT = MC / 16;
  for (int it = 0; it < NIT; ++it) {
    const int buf = it & 1;
    __syncthreads();                       // staged(buf) visible; buf^1 free
    if (it + 1 < NIT) stage(buf ^ 1, (it + 1) * 16);

    const unsigned short* kp = &Ks[buf][lane16 * 72 + laneHi * 16];
    v16bf b0 = load_frag16(kp);            // K rows 0..31
    v16bf b1 = load_frag16(kp + 32);       // K rows 32..63

    v8f c = {};
    c = __builtin_amdgcn_wmma_f32_16x16x32_bf16(false, a0, false, b0,
                                                (short)0, c, false, false);
    c = __builtin_amdgcn_wmma_f32_16x16x32_bf16(false, a1, false, b1,
                                                (short)0, c, false, false);

    const int n  = it * 16 + lane16;
    const float gx = (2.0f * (float)(n & (WC - 1)) + 1.0f) / (float)WC - 1.0f;
    const float gy = (2.0f * (float)(n >> 6)       + 1.0f) / (float)HC - 1.0f;
#pragma unroll
    for (int r = 0; r < 8; ++r) {
      float s  = c[r];
      float d  = s - mrow[r];              // +inf on first hit
      float e  = __expf(-fabsf(d));        // single exp: either p or scale
      float p  = (d <= 0.f) ? e : 1.f;
      float sc = (d <= 0.f) ? 1.f : e;
      mrow[r] = fmaxf(mrow[r], s);
      lrow[r] = lrow[r] * sc + p;
      ax[r]   = ax[r]   * sc + p * gx;
      ay[r]   = ay[r]   * sc + p * gy;
    }
  }

  // exact flash merge across the 16 lanes of each row group
#pragma unroll
  for (int r = 0; r < 8; ++r) {
    float m = mrow[r], l = lrow[r], x = ax[r], y = ay[r];
#pragma unroll
    for (int msk = 1; msk <= 8; msk <<= 1) {
      float m2 = __shfl_xor(m, msk);
      float l2 = __shfl_xor(l, msk);
      float x2 = __shfl_xor(x, msk);
      float y2 = __shfl_xor(y, msk);
      float mn = fmaxf(m, m2);
      float ea = __expf(m - mn);
      float eb = __expf(m2 - mn);
      l = l * ea + l2 * eb;
      x = x * ea + x2 * eb;
      y = y * ea + y2 * eb;
      m = mn;
    }
    if (lane16 == 0) {
      int mi = m0 + r + 8 * laneHi;
      float inv = 1.0f / l;
      warp[(size_t)b * 2 * MC + mi]      = x * inv;
      warp[(size_t)b * 2 * MC + MC + mi] = y * inv;
    }
  }
}

// ---------------------------------------------------------------------------
// Kernel 2: coarse concat pack -> bf16 (130 ch)
// ---------------------------------------------------------------------------
__global__ void pack_coarse_kernel(const float* __restrict__ f0c,
                                   const float* __restrict__ f1c,
                                   const float* __restrict__ warp,
                                   unsigned short* __restrict__ x0) {
  int t = blockIdx.x * blockDim.x + threadIdx.x;
  if (t >= NB * MC) return;
  int b = t / MC, pix = t - b * MC;
  float gxv = warp[(size_t)b * 2 * MC + pix];
  float gyv = warp[(size_t)b * 2 * MC + MC + pix];
  float x = (gxv + 1.0f) * (WC * 0.5f) - 0.5f;
  float y = (gyv + 1.0f) * (HC * 0.5f) - 0.5f;
  float xf = floorf(x), yf = floorf(y);
  float wx = x - xf, wy = y - yf;
  int xi = (int)xf, yi = (int)yf;
  float w00 = (1.f - wx) * (1.f - wy), w01 = wx * (1.f - wy);
  float w10 = (1.f - wx) * wy,         w11 = wx * wy;
  int x0c = iclamp(xi, 0, WC - 1),     x1c = iclamp(xi + 1, 0, WC - 1);
  int y0c = iclamp(yi, 0, HC - 1),     y1c = iclamp(yi + 1, 0, HC - 1);
  float v00m = (xi >= 0 && xi < WC && yi >= 0 && yi < HC) ? 1.f : 0.f;
  float v01m = (xi + 1 >= 0 && xi + 1 < WC && yi >= 0 && yi < HC) ? 1.f : 0.f;
  float v10m = (xi >= 0 && xi < WC && yi + 1 >= 0 && yi + 1 < HC) ? 1.f : 0.f;
  float v11m = (xi + 1 >= 0 && xi + 1 < WC && yi + 1 >= 0 && yi + 1 < HC) ? 1.f : 0.f;

  const float* f0b = f0c + (size_t)b * CC * MC;
  const float* f1b = f1c + (size_t)b * CC * MC;
  unsigned short* xb = x0 + (size_t)b * 130 * MC;
  for (int c = 0; c < CC; ++c) {
    const float* im = f1b + (size_t)c * MC;
    float v = im[y0c * WC + x0c] * v00m * w00 + im[y0c * WC + x1c] * v01m * w01 +
              im[y1c * WC + x0c] * v10m * w10 + im[y1c * WC + x1c] * v11m * w11;
    xb[(size_t)c * MC + pix]        = f2bfbits(f0b[(size_t)c * MC + pix]);
    xb[(size_t)(CC + c) * MC + pix] = f2bfbits(v);
  }
  xb[(size_t)128 * MC + pix] = f2bfbits(gxv);
  xb[(size_t)129 * MC + pix] = f2bfbits(gyv);
}

// ---------------------------------------------------------------------------
// Kernel 3: 3x3 SAME conv, implicit GEMM on WMMA bf16.
// Block=256 (8 waves) -> 32 px x 64 cout tile; K padded to mult of 32 (zero
// weights) so the loop is branch-free; pixel geometry hoisted out of K-loop.
// As: [px][32k linear], rows padded to 40 shorts.
// Bs: [cout][32k linear], rows padded to 40 shorts. Fragments = 2x b128.
// ---------------------------------------------------------------------------
__global__ __launch_bounds__(256) void conv3x3_wmma_kernel(
    const unsigned short* __restrict__ in,   // (B,Cin,H,W) bf16 bits
    const unsigned short* __restrict__ wt,   // (Cout, Kpad) bf16 bits, 0-padded
    float* __restrict__ out,                 // (B,Cout,H,W) fp32
    int Cin, int Cout, int H, int W, int Kpad) {
  __shared__ unsigned short As[32 * 40];
  __shared__ unsigned short Bs[64 * 40];
  const int HW = H * W;
  const int lane = threadIdx.x & 31, wave = threadIdx.x >> 5;
  const int laneHi = lane >> 4, lane16 = lane & 15;
  const int pr = wave >> 2, cb = wave & 3;
  const int pBase  = blockIdx.x * 32;
  const int coBase = blockIdx.y * 64;

  // ---- A staging geometry (fixed pixel per thread, 4 consecutive k) ----
  const int s_pl  = threadIdx.x >> 3;        // 0..31
  const int s_kl0 = (threadIdx.x & 7) * 4;   // 0,4,..,28 (aligned 4-run)
  const int s_p   = pBase + s_pl;
  const int s_b   = s_p / HW;
  const int s_hw  = s_p - s_b * HW;
  const int s_h   = s_hw / W;
  const int s_w   = s_hw - s_h * W;
  const unsigned short* s_inb = in + (size_t)s_b * Cin * HW;
  const int s_pos0 = s_kl0;                  // 4-runs stay contiguous

  // ---- B staging geometry (fixed cout per thread, 8 consecutive k) ----
  const int w_col = threadIdx.x >> 2;        // 0..63
  const int w_kl0 = (threadIdx.x & 3) * 8;   // 0,8,16,24
  const unsigned short* w_row = wt + (size_t)(coBase + w_col) * Kpad + w_kl0;

  v8f acc = {};
  for (int k0 = 0; k0 < Kpad; k0 += 32) {
    __syncthreads();
    // A tile (im2col, branch-free thanks to zero-padded weights)
    unsigned short av[4];
#pragma unroll
    for (int j = 0; j < 4; ++j) {
      int k = k0 + s_kl0 + j;
      int cin = k / 9;
      int r9 = k - cin * 9;
      if (cin >= Cin) cin = Cin - 1;          // pad region: value x 0-weight
      int kh = r9 / 3, kw = r9 - kh * 3;
      int hi = s_h + kh - 1, wi = s_w + kw - 1;
      unsigned short v = 0;
      if (hi >= 0 && hi < H && wi >= 0 && wi < W)
        v = s_inb[(size_t)cin * HW + hi * W + wi];
      av[j] = v;
    }
    u32x2 apk;
    apk.x = (unsigned)av[0] | ((unsigned)av[1] << 16);
    apk.y = (unsigned)av[2] | ((unsigned)av[3] << 16);
    *(u32x2*)&As[s_pl * 40 + s_pos0] = apk;
    // B tile (16B global load -> 16B LDS store)
    *(u32x4*)&Bs[w_col * 40 + w_kl0] = *(const u32x4*)(w_row + k0);
    __syncthreads();

    if (k0 + 32 < Kpad)                      // global_prefetch_b8 of next weights
      __builtin_prefetch(w_row + k0 + 32, 0, 0);

    v16bf a  = load_frag16(&As[(pr * 16 + lane16) * 40 + laneHi * 16]);
    v16bf bb = load_frag16(&Bs[(cb * 16 + lane16) * 40 + laneHi * 16]);
    acc = __builtin_amdgcn_wmma_f32_16x16x32_bf16(false, a, false, bb,
                                                  (short)0, acc, false, false);
  }

  // contiguous 32B per-lane store of the 8 accumulator rows
  const int co = coBase + cb * 16 + lane16;
  const int p0 = pBase + pr * 16 + 8 * laneHi;   // rows r -> pixel p0 + r
  const int ob = p0 / HW;
  const int ohw = p0 - ob * HW;
  float* op = out + (size_t)(ob * Cout + co) * HW + ohw;
#pragma unroll
  for (int r = 0; r < 8; ++r) op[r] = acc[r];
}

// ---------------------------------------------------------------------------
// BN: per-channel mean / inv-std over (B,H,W), then normalize+ReLU -> bf16.
// ---------------------------------------------------------------------------
__global__ __launch_bounds__(256) void bn_stats_kernel(
    const float* __restrict__ x, float* __restrict__ mean,
    float* __restrict__ istd, int C, int HW) {
  __shared__ float ssum[256], ssq[256];
  int c = blockIdx.x;
  float s = 0.f, q = 0.f;
  for (int b = 0; b < NB; ++b) {
    const float* xb = x + (size_t)(b * C + c) * HW;
    for (int i = threadIdx.x; i < HW; i += 256) {
      float v = xb[i];
      s += v; q += v * v;
    }
  }
  ssum[threadIdx.x] = s; ssq[threadIdx.x] = q;
  __syncthreads();
  for (int o = 128; o > 0; o >>= 1) {
    if (threadIdx.x < o) {
      ssum[threadIdx.x] += ssum[threadIdx.x + o];
      ssq[threadIdx.x]  += ssq[threadIdx.x + o];
    }
    __syncthreads();
  }
  if (threadIdx.x == 0) {
    float n = (float)(NB * HW);
    float m = ssum[0] / n;
    float var = ssq[0] / n - m * m;
    mean[c] = m;
    istd[c] = rsqrtf(var + 1e-5f);
  }
}

// grid = (ceil(HW/256), C, B): no per-element div/mod
__global__ void bn_relu_bf16_kernel(const float* __restrict__ x,
                                    const float* __restrict__ mean,
                                    const float* __restrict__ istd,
                                    unsigned short* __restrict__ y, int HW) {
  int hw = blockIdx.x * blockDim.x + threadIdx.x;
  if (hw >= HW) return;
  int c = blockIdx.y, C = gridDim.y, b = blockIdx.z;
  size_t i = (size_t)(b * C + c) * HW + hw;
  float v = (x[i] - mean[c]) * istd[c];
  y[i] = f2bfbits(fmaxf(v, 0.0f));
}

// ---------------------------------------------------------------------------
// 1x1 conv heads
// ---------------------------------------------------------------------------
__global__ void head_coarse_kernel(const unsigned short* __restrict__ act,
                                   const float* __restrict__ w5,
                                   const float* __restrict__ b5,
                                   const float* __restrict__ warp,
                                   float* __restrict__ out /* (B,3,MC) */) {
  int t = blockIdx.x * blockDim.x + threadIdx.x;
  if (t >= NB * MC) return;
  int b = t / MC, pix = t - b * MC;
  const unsigned short* ab = act + (size_t)b * 256 * MC;
  float a0 = 0.f, a1 = 0.f, a2 = 0.f;
  for (int c = 0; c < 256; ++c) {
    float v = bf2f(ab[(size_t)c * MC + pix]);
    a0 += v * w5[c]; a1 += v * w5[256 + c]; a2 += v * w5[512 + c];
  }
  float bx = warp[(size_t)b * 2 * MC + pix];
  float by = warp[(size_t)b * 2 * MC + MC + pix];
  out[(size_t)b * 3 * MC + pix]          = bx + a0 + b5[0];
  out[(size_t)b * 3 * MC + MC + pix]     = by + a1 + b5[1];
  out[(size_t)b * 3 * MC + 2 * MC + pix] =      a2 + b5[2];
}

__global__ void head_fine_kernel(const unsigned short* __restrict__ act,
                                 const float* __restrict__ w5,
                                 const float* __restrict__ b5,
                                 const float* __restrict__ up,
                                 float* __restrict__ out /* (B,3,MF) */) {
  int t = blockIdx.x * blockDim.x + threadIdx.x;
  if (t >= NB * MF) return;
  int b = t / MF, pix = t - b * MF;
  const unsigned short* ab = act + (size_t)b * 64 * MF;
  float a0 = 0.f, a1 = 0.f, a2 = 0.f;
  for (int c = 0; c < 64; ++c) {
    float v = bf2f(ab[(size_t)c * MF + pix]);
    a0 += v * w5[c]; a1 += v * w5[64 + c]; a2 += v * w5[128 + c];
  }
  size_t base = (size_t)b * 3 * MF + pix;
  out[base]          = up[base]          + a0 + b5[0];
  out[base + MF]     = up[base + MF]     + a1 + b5[1];
  out[base + 2 * MF] = up[base + 2 * MF] + a2 + b5[2];
}

// ---------------------------------------------------------------------------
// Bilinear 2x upsample (half-pixel centers, edge clamp)
// ---------------------------------------------------------------------------
__global__ void upsample2x_kernel(const float* __restrict__ in,
                                  float* __restrict__ out) {
  int t = blockIdx.x * blockDim.x + threadIdx.x;
  if (t >= NB * 3 * MF) return;
  int pix = t % MF, bc = t / MF;
  int yo = pix >> 7, xo = pix & (WF - 1);
  float sx = (xo + 0.5f) * 0.5f - 0.5f;
  float sy = (yo + 0.5f) * 0.5f - 0.5f;
  float xf = floorf(sx), yf = floorf(sy);
  float wx = sx - xf, wy = sy - yf;
  int x0 = iclamp((int)xf, 0, WC - 1), x1 = iclamp((int)xf + 1, 0, WC - 1);
  int y0 = iclamp((int)yf, 0, HC - 1), y1 = iclamp((int)yf + 1, 0, HC - 1);
  const float* ib = in + (size_t)bc * MC;
  out[t] = ib[y0 * WC + x0] * (1.f - wx) * (1.f - wy) +
           ib[y0 * WC + x1] * wx * (1.f - wy) +
           ib[y1 * WC + x0] * (1.f - wx) * wy +
           ib[y1 * WC + x1] * wx * wy;
}

// ---------------------------------------------------------------------------
// Fine concat pack -> bf16 (50 ch)
// ---------------------------------------------------------------------------
__global__ void pack_fine_kernel(const float* __restrict__ f0f,
                                 const float* __restrict__ f1f,
                                 const float* __restrict__ up,
                                 unsigned short* __restrict__ y0) {
  int t = blockIdx.x * blockDim.x + threadIdx.x;
  if (t >= NB * MF) return;
  int b = t / MF, pix = t - b * MF;
  float gxv = up[(size_t)b * 3 * MF + pix];
  float gyv = up[(size_t)b * 3 * MF + MF + pix];
  float x = (gxv + 1.0f) * (WF * 0.5f) - 0.5f;
  float y = (gyv + 1.0f) * (HF * 0.5f) - 0.5f;
  float xf = floorf(x), yf = floorf(y);
  float wx = x - xf, wy = y - yf;
  int xi = (int)xf, yi = (int)yf;
  float w00 = (1.f - wx) * (1.f - wy), w01 = wx * (1.f - wy);
  float w10 = (1.f - wx) * wy,         w11 = wx * wy;
  int x0c = iclamp(xi, 0, WF - 1),     x1c = iclamp(xi + 1, 0, WF - 1);
  int y0c = iclamp(yi, 0, HF - 1),     y1c = iclamp(yi + 1, 0, HF - 1);
  float v00m = (xi >= 0 && xi < WF && yi >= 0 && yi < HF) ? 1.f : 0.f;
  float v01m = (xi + 1 >= 0 && xi + 1 < WF && yi >= 0 && yi < HF) ? 1.f : 0.f;
  float v10m = (xi >= 0 && xi < WF && yi + 1 >= 0 && yi + 1 < HF) ? 1.f : 0.f;
  float v11m = (xi + 1 >= 0 && xi + 1 < WF && yi + 1 >= 0 && yi + 1 < HF) ? 1.f : 0.f;

  const float* f0b = f0f + (size_t)b * CF * MF;
  const float* f1b = f1f + (size_t)b * CF * MF;
  unsigned short* yb = y0 + (size_t)b * 50 * MF;
  for (int c = 0; c < CF; ++c) {
    const float* im = f1b + (size_t)c * MF;
    float v = im[y0c * WF + x0c] * v00m * w00 + im[y0c * WF + x1c] * v01m * w01 +
              im[y1c * WF + x0c] * v10m * w10 + im[y1c * WF + x1c] * v11m * w11;
    yb[(size_t)c * MF + pix]        = f2bfbits(f0b[(size_t)c * MF + pix]);
    yb[(size_t)(CF + c) * MF + pix] = f2bfbits(v);
  }
  yb[(size_t)48 * MF + pix] = f2bfbits(gxv);
  yb[(size_t)49 * MF + pix] = f2bfbits(gyv);
}

// weights fp32 (Cout,Ktot) -> bf16 (Cout,Kpad) zero-padded in K
__global__ void wt_pack_kernel(const float* __restrict__ src,
                               unsigned short* __restrict__ dst,
                               int Cout, int Ktot, int Kpad) {
  int i = blockIdx.x * blockDim.x + threadIdx.x;
  if (i >= Cout * Kpad) return;
  int co = i / Kpad, k = i - co * Kpad;
  dst[i] = (k < Ktot) ? f2bfbits(src[(size_t)co * Ktot + k]) : (unsigned short)0;
}

// ---------------------------------------------------------------------------
extern "C" void kernel_launch(void* const* d_in, const int* in_sizes, int n_in,
                              void* d_out, int out_size, void* d_ws, size_t ws_size,
                              hipStream_t stream) {
  (void)in_sizes; (void)n_in; (void)out_size; (void)ws_size;
  const float* f0c = (const float*)d_in[0];
  const float* f1c = (const float*)d_in[1];
  const float* f0f = (const float*)d_in[2];
  const float* f1f = (const float*)d_in[3];
  const float* cw1 = (const float*)d_in[4];
  const float* cw2 = (const float*)d_in[5];
  const float* cw3 = (const float*)d_in[6];
  const float* cw4 = (const float*)d_in[7];
  const float* cw5 = (const float*)d_in[8];
  const float* cb5 = (const float*)d_in[9];
  const float* fw1 = (const float*)d_in[10];
  const float* fw2 = (const float*)d_in[11];
  const float* fw3 = (const float*)d_in[12];
  const float* fw4 = (const float*)d_in[13];
  const float* fw5 = (const float*)d_in[14];
  const float* fb5 = (const float*)d_in[15];

  char* ws = (char*)d_ws;
  size_t off = 0;
  auto carve = [&](size_t bytes) -> void* {
    void* p = ws + off;
    off += (bytes + 255) & ~(size_t)255;
    return p;
  };
  const int KC1 = 1184, KC = 2304, KF1 = 480, KF = 576;  // padded K sizes
  float*          warp    = (float*)carve((size_t)NB * 2 * MC * 4);
  unsigned short* x0      = (unsigned short*)carve((size_t)NB * 130 * MC * 2);
  float*          convOut = (float*)carve((size_t)NB * 256 * MC * 4);
  unsigned short* actA    = (unsigned short*)carve((size_t)NB * 256 * MC * 2);
  unsigned short* actB    = (unsigned short*)carve((size_t)NB * 256 * MC * 2);
  float*          mean    = (float*)carve(256 * 4);
  float*          istd    = (float*)carve(256 * 4);
  float*          up      = (float*)carve((size_t)NB * 3 * MF * 4);
  unsigned short* y0      = (unsigned short*)carve((size_t)NB * 50 * MF * 2);
  unsigned short* cw1b    = (unsigned short*)carve((size_t)256 * KC1 * 2);
  unsigned short* cw2b    = (unsigned short*)carve((size_t)256 * KC * 2);
  unsigned short* cw3b    = (unsigned short*)carve((size_t)256 * KC * 2);
  unsigned short* cw4b    = (unsigned short*)carve((size_t)256 * KC * 2);
  unsigned short* fw1b    = (unsigned short*)carve((size_t)64 * KF1 * 2);
  unsigned short* fw2b    = (unsigned short*)carve((size_t)64 * KF * 2);
  unsigned short* fw3b    = (unsigned short*)carve((size_t)64 * KF * 2);
  unsigned short* fw4b    = (unsigned short*)carve((size_t)64 * KF * 2);

  auto pack = [&](const float* src, unsigned short* dst, int Cout, int Ktot, int Kpad) {
    int n = Cout * Kpad;
    wt_pack_kernel<<<(n + 255) / 256, 256, 0, stream>>>(src, dst, Cout, Ktot, Kpad);
  };
  pack(cw1, cw1b, 256, 130 * 9, KC1);
  pack(cw2, cw2b, 256, 256 * 9, KC);
  pack(cw3, cw3b, 256, 256 * 9, KC);
  pack(cw4, cw4b, 256, 256 * 9, KC);
  pack(fw1, fw1b, 64, 50 * 9, KF1);
  pack(fw2, fw2b, 64, 64 * 9, KF);
  pack(fw3, fw3b, 64, 64 * 9, KF);
  pack(fw4, fw4b, 64, 64 * 9, KF);

  // 1) flash-attention correlation -> coarse warp field
  corr_softmax_warp_kernel<<<dim3(MC / 64, NB), 128, 0, stream>>>(f0c, f1c, warp);

  // 2) coarse concat (bf16)
  pack_coarse_kernel<<<(NB * MC + 255) / 256, 256, 0, stream>>>(f0c, f1c, warp, x0);

  // 3) conv stacks (WMMA implicit GEMM + BN/ReLU)
  auto conv_bn = [&](const unsigned short* in, const unsigned short* w,
                     unsigned short* act, int Cin, int Cout, int H, int W, int Kpad) {
    int HW = H * W;
    dim3 grid((NB * HW) / 32, Cout / 64);
    conv3x3_wmma_kernel<<<grid, 256, 0, stream>>>(in, w, convOut, Cin, Cout, H, W, Kpad);
    bn_stats_kernel<<<Cout, 256, 0, stream>>>(convOut, mean, istd, Cout, HW);
    dim3 g2((HW + 255) / 256, Cout, NB);
    bn_relu_bf16_kernel<<<g2, 256, 0, stream>>>(convOut, mean, istd, act, HW);
  };
  conv_bn(x0,   cw1b, actA, 130, 256, HC, WC, KC1);
  conv_bn(actA, cw2b, actB, 256, 256, HC, WC, KC);
  conv_bn(actB, cw3b, actA, 256, 256, HC, WC, KC);
  conv_bn(actA, cw4b, actB, 256, 256, HC, WC, KC);

  float* outC = (float*)d_out;                 // (B,3,64,64)
  float* outF = outC + (size_t)NB * 3 * MC;    // (B,3,128,128)

  // 4) coarse head
  head_coarse_kernel<<<(NB * MC + 255) / 256, 256, 0, stream>>>(actB, cw5, cb5,
                                                                warp, outC);
  // 5) upsample
  upsample2x_kernel<<<(NB * 3 * MF + 255) / 256, 256, 0, stream>>>(outC, up);
  // 6) fine concat
  pack_fine_kernel<<<(NB * MF + 255) / 256, 256, 0, stream>>>(f0f, f1f, up, y0);
  // 7) fine conv stack
  conv_bn(y0,   fw1b, actA, 50, 64, HF, WF, KF1);
  conv_bn(actA, fw2b, actB, 64, 64, HF, WF, KF);
  conv_bn(actB, fw3b, actA, 64, 64, HF, WF, KF);
  conv_bn(actA, fw4b, actB, 64, 64, HF, WF, KF);
  // 8) fine head
  head_fine_kernel<<<(NB * MF + 255) / 256, 256, 0, stream>>>(actB, fw5, fb5,
                                                              up, outF);
}
